// GraphCast_89799176225499
// MI455X (gfx1250) — compile-verified, matrix operand order checked by reading, not run on hardware
//
#include <hip/hip_runtime.h>
#include <hip/hip_bf16.h>

// ---------------------------------------------------------------------------
// GraphCast forward for MI455X (gfx1250), wave32 + WMMA bf16 16x16x32.
//
// Param flattening assumption (jax pytree: dicts by sorted key, tuples in
// order, _lin_p = (w,b), _mlp3_p keys sorted = l1,l2,l3,ln_b,ln_w):
//   d_in[10..]: emb_g(w,b) emb_g2m emb_m emb_m2g emb_me |
//   g2m_edge[20..27] g2m_grid[28..33] g2m_node[34..41] head[42..47]
//   m2g_edge[48..55] m2g_node[56..63] me_edge1[64..71] me_edge2[72..79]
//   me_node1[80..87] me_node2[88..95]
// ---------------------------------------------------------------------------

typedef __attribute__((ext_vector_type(16))) __bf16 v16bf;
typedef __attribute__((ext_vector_type(8)))  __bf16 v8bf;
typedef __attribute__((ext_vector_type(8)))  float  v8f;

#define BS_   2
#define GNUM_ 25921
#define MNUM_ 1681
#define ME_E  4722
#define G2M_E 36386
#define M2G_E 56672

struct GSrc {
  const float* base;   // per-batch block of rows*kw, row-major
  const int*   idx;    // per-edge gather index (nullptr => identity)
  int rows;            // rows per batch in source
  int kw;              // feature width of this concat segment (multiple of 32)
};

// ---------------------------------------------------------------------------
// Fused gather-concat GEMM:  out[r,:] = act( concat(s0,s1,s2)[r,:] @ W + b )
// - A staged f32->bf16 into LDS (gather fused, concat never materialized)
// - B (K x N, f32) staged transposed (n-major) -> frag loads = ds_load_b128
// - double-buffered LDS; staging split: next tile's GLOBAL loads issue before
//   the WMMA group (latency hidden under matrix ops), cvt + LDS stores issue
//   after it (pre-WMMA dscnt waits cover only the 10 fragment loads)
// ---------------------------------------------------------------------------

// phase 1: uniform segment select + global loads into sf0/sf1/swv
#define STAGE_LOAD(KK)                                                        \
  {                                                                           \
    const int kk_ = (KK);                                                     \
    const int sis_ = (kk_ >= c0 ? 1 : 0) + (kk_ >= c1 ? 1 : 0);               \
    const int kl_ = kk_ - (sis_ > 0 ? c0 : 0) - (sis_ > 1 ? (c1 - c0) : 0);   \
    const float* sbp_ = (sis_ == 0) ? sb0 : (sis_ == 1 ? sb1 : sb2);          \
    unsigned long long off_ = roff[sis_][arl];                                \
    if (off_ != ~0ULL) {                                                      \
      const float* p_ = sbp_ + off_ + (unsigned)(kl_ + aq);                   \
      sf0 = *(const float4*)(p_);                                             \
      sf1 = *(const float4*)(p_ + 4);                                         \
    } else {                                                                  \
      sf0 = f4z_; sf1 = f4z_;                                                 \
    }                                                                         \
    const float* wp_ = Wm + (size_t)(kk_ + bkq) * N + bn;                     \
    if (bn < N) {                                                             \
      _Pragma("unroll")                                                       \
      for (int i_ = 0; i_ < 16; ++i_) swv[i_] = wp_[(size_t)i_ * N];          \
    } else {                                                                  \
      _Pragma("unroll")                                                       \
      for (int i_ = 0; i_ < 16; ++i_) swv[i_] = 0.f;                          \
    }                                                                         \
  }

// phase 2: f32->bf16 convert + LDS stores
#define STAGE_STORE(BUF)                                                      \
  {                                                                           \
    __bf16* da_ = &la[BUF][arl][aq];                                          \
    da_[0] = (__bf16)sf0.x; da_[1] = (__bf16)sf0.y;                           \
    da_[2] = (__bf16)sf0.z; da_[3] = (__bf16)sf0.w;                           \
    da_[4] = (__bf16)sf1.x; da_[5] = (__bf16)sf1.y;                           \
    da_[6] = (__bf16)sf1.z; da_[7] = (__bf16)sf1.w;                           \
    __bf16* db_ = &lb[BUF][bnl][bkq];                                         \
    _Pragma("unroll")                                                         \
    for (int i_ = 0; i_ < 16; ++i_) db_[i_] = (__bf16)swv[i_];                \
  }

__global__ __launch_bounds__(256, 2)
void wmma_mlp_kernel(GSrc s0, GSrc s1, GSrc s2,
                     const float* __restrict__ Wm, const float* __restrict__ bias,
                     float* __restrict__ out, int M, int N, int K,
                     int rowsPerBatch, int ldo, int act) {
  constexpr int BM = 64, BN = 128, BK = 32, LD = BK + 8;  // 80B row stride
  __shared__ __bf16 la[2][BM][LD];
  __shared__ __bf16 lb[2][BN][LD];
  __shared__ unsigned long long roff[3][BM];

  // struct fields -> plain locals (uniform values stay in SGPRs)
  const float* sb0 = s0.base; const int* ix0 = s0.idx;
  const int sr0 = s0.rows, sk0 = s0.kw;
  const float* sb1 = s1.base; const int* ix1 = s1.idx;
  const int sr1 = s1.rows, sk1 = s1.kw;
  const float* sb2 = s2.base; const int* ix2 = s2.idx;
  const int sr2 = s2.rows, sk2 = s2.kw;
  const int c0 = sk0;
  const int c1 = sk0 + sk1;

  const int tid = threadIdx.x;
  const int m0 = blockIdx.x * BM;
  const int n0 = blockIdx.y * BN;

  if (tid < BM) {
    int r = m0 + tid;
    bool valid = (r < M);
    int b = 0, e = 0;
    if (valid) { b = r / rowsPerBatch; e = r - b * rowsPerBatch; }
    unsigned long long o0 = ~0ULL, o1 = ~0ULL, o2 = ~0ULL;
    if (valid && sk0) {
      int sel = ix0 ? ix0[e] : e;
      o0 = ((unsigned long long)((long long)b * sr0 + sel)) * (unsigned long long)sk0;
    }
    if (valid && sk1) {
      int sel = ix1 ? ix1[e] : e;
      o1 = ((unsigned long long)((long long)b * sr1 + sel)) * (unsigned long long)sk1;
    }
    if (valid && sk2) {
      int sel = ix2 ? ix2[e] : e;
      o2 = ((unsigned long long)((long long)b * sr2 + sel)) * (unsigned long long)sk2;
    }
    roff[0][tid] = o0; roff[1][tid] = o1; roff[2][tid] = o2;
  }

  const int lane = tid & 31;
  const int wv = tid >> 5;
  const int wm = wv & 3;        // wave row tile (16 rows each)
  const int wn = wv >> 2;       // wave col tile (64 cols each)
  const int ml = lane & 15;
  const int kh = lane >> 4;     // K-half selector per ISA 7.12.2

  const int arl = tid >> 2;              // A-stage row 0..63
  const int aq  = (tid & 3) * 8;         // A-stage k offset (8 floats each)
  const int bnl = tid & 127;             // B-stage column
  const int bkq = (tid >> 7) * 16;       // B-stage k offset (16 each)
  const int bn  = n0 + bnl;

  v8f acc[4];
#pragma unroll
  for (int j = 0; j < 4; ++j)
#pragma unroll
    for (int v = 0; v < 8; ++v) acc[j][v] = 0.f;

  const float4 f4z_ = {0.f, 0.f, 0.f, 0.f};
  float4 sf0 = f4z_, sf1 = f4z_;
  float swv[16];

  const int ktiles = K / BK;
  __syncthreads();             // roff visible
  STAGE_LOAD(0);
  STAGE_STORE(0);

  union V16 { v16bf v; v8bf h[2]; };
  const int ar = wm * 16 + ml;
  const int bc0 = wn * 64 + ml;

  for (int t = 0; t < ktiles; ++t) {
    __syncthreads();           // stage(t) done; prior tile's frag reads done
    const int buf = t & 1;
    const bool hn = (t + 1 < ktiles);
    // fragment loads: batched ds_load_b128, independent registers
    V16 af, bf0, bf1, bf2, bf3;
    af.h[0]  = *(const v8bf*)&la[buf][ar][kh * 8];
    af.h[1]  = *(const v8bf*)&la[buf][ar][16 + kh * 8];
    bf0.h[0] = *(const v8bf*)&lb[buf][bc0     ][kh * 8];
    bf0.h[1] = *(const v8bf*)&lb[buf][bc0     ][16 + kh * 8];
    bf1.h[0] = *(const v8bf*)&lb[buf][bc0 + 16][kh * 8];
    bf1.h[1] = *(const v8bf*)&lb[buf][bc0 + 16][16 + kh * 8];
    bf2.h[0] = *(const v8bf*)&lb[buf][bc0 + 32][kh * 8];
    bf2.h[1] = *(const v8bf*)&lb[buf][bc0 + 32][16 + kh * 8];
    bf3.h[0] = *(const v8bf*)&lb[buf][bc0 + 48][kh * 8];
    bf3.h[1] = *(const v8bf*)&lb[buf][bc0 + 48][16 + kh * 8];
    // issue next tile's global loads now (latency hidden under WMMAs)
    if (hn) STAGE_LOAD((t + 1) * BK);
    acc[0] = __builtin_amdgcn_wmma_f32_16x16x32_bf16(false, af.v, false, bf0.v,
                                                     (short)0, acc[0], false, false);
    acc[1] = __builtin_amdgcn_wmma_f32_16x16x32_bf16(false, af.v, false, bf1.v,
                                                     (short)0, acc[1], false, false);
    acc[2] = __builtin_amdgcn_wmma_f32_16x16x32_bf16(false, af.v, false, bf2.v,
                                                     (short)0, acc[2], false, false);
    acc[3] = __builtin_amdgcn_wmma_f32_16x16x32_bf16(false, af.v, false, bf3.v,
                                                     (short)0, acc[3], false, false);
    // convert + LDS stores after the WMMA group
    if (hn) STAGE_STORE(buf ^ 1);
  }

  // ---- epilogue: bias + optional SiLU, guarded stores ----
#pragma unroll
  for (int j = 0; j < 4; ++j) {
    int col = n0 + wn * 64 + j * 16 + ml;
    if (col < N) {
      float bb = bias[col];
#pragma unroll
      for (int v = 0; v < 8; ++v) {
        int row = m0 + wm * 16 + kh * 8 + v;   // C layout: VGPR v -> M=v(+8*kh)
        if (row < M) {
          float x = acc[j][v] + bb;
          if (act) x = x / (1.f + __expf(-x));
          out[(size_t)row * ldo + col] = x;
        }
      }
    }
  }
}

// ---------------------------------------------------------------------------
// Grid feature embedding: K=142 (140 transposed channels + gs + lsm), N=256.
// ---------------------------------------------------------------------------
__global__ __launch_bounds__(256)
void emb_grid_kernel(const float* __restrict__ gx, const float* __restrict__ gs,
                     const float* __restrict__ lsm, const float* __restrict__ Wm,
                     const float* __restrict__ bias, float* __restrict__ out) {
  __shared__ float f[144];
  const int bp = blockIdx.x;                 // b*GNUM + p
  const int b = bp / GNUM_, p = bp - b * GNUM_;
  const int t = threadIdx.x;
  if (t < 140)       f[t] = gx[((size_t)(b * 140 + t)) * GNUM_ + p];
  else if (t == 140) f[t] = gs[bp];
  else if (t == 141) f[t] = lsm[bp];
  __syncthreads();
  float acc = bias[t];
  for (int k = 0; k < 142; ++k) acc += f[k] * Wm[k * 256 + t];
  out[(size_t)bp * 256 + t] = acc;
}

// Tiny-K embeddings (K=3/4), replicated over `copies` batches.
__global__ void small_emb_kernel(const float* __restrict__ x,
                                 const float* __restrict__ Wm,
                                 const float* __restrict__ bias,
                                 float* __restrict__ out,
                                 int rows, int K, int copies) {
  long long t = (long long)blockIdx.x * 256 + threadIdx.x;
  long long total = (long long)rows * 256;
  if (t >= total) return;
  int r = (int)(t >> 8), n = (int)(t & 255);
  float acc = bias[n];
  for (int k = 0; k < K; ++k) acc += x[(size_t)r * K + k] * Wm[k * 256 + n];
  for (int c = 0; c < copies; ++c) out[(size_t)c * total + t] = acc;
}

__global__ void zero_kernel(float* p, long long n) {
  long long t = (long long)blockIdx.x * 256 + threadIdx.x;
  if (t < n) p[t] = 0.f;
}

// agg[b, col[e], :] += src[b, e, :]
__global__ void scatter_add_kernel(float* __restrict__ dst,
                                   const float* __restrict__ src,
                                   const int* __restrict__ col,
                                   int E, int dstRows) {
  long long t = (long long)blockIdx.x * 256 + threadIdx.x;
  long long total = (long long)BS_ * E * 256;
  if (t >= total) return;
  int n = (int)(t & 255);
  long long re = t >> 8;
  int b = (int)(re / E);
  int e = (int)(re - (long long)b * E);
  atomicAdd(&dst[(((size_t)b * dstRows) + col[e]) * 256 + n], src[t]);
}

// Per-batch global sum / sumsq over rowsPerBatch*256 elems (LN over (N,D)).
__global__ __launch_bounds__(256)
void ln_stats_kernel(const float* __restrict__ x, int rowsPerBatch,
                     double* __restrict__ stats) {
  __shared__ float r1[256], r2[256];
  const long long cnt = (long long)rowsPerBatch * 256;
  const int b = blockIdx.y;
  const float* xb = x + (size_t)b * cnt;
  float a1 = 0.f, a2 = 0.f;
  for (long long i = (long long)blockIdx.x * 256 + threadIdx.x; i < cnt;
       i += (long long)gridDim.x * 256) {
    float v = xb[i]; a1 += v; a2 += v * v;
  }
  r1[threadIdx.x] = a1; r2[threadIdx.x] = a2;
  __syncthreads();
  for (int s = 128; s > 0; s >>= 1) {
    if (threadIdx.x < s) { r1[threadIdx.x] += r1[threadIdx.x + s];
                           r2[threadIdx.x] += r2[threadIdx.x + s]; }
    __syncthreads();
  }
  if (threadIdx.x == 0) {
    atomicAdd(&stats[2 * b],     (double)r1[0]);
    atomicAdd(&stats[2 * b + 1], (double)r2[0]);
  }
}

// state += (h - mu) * rsqrt(var+eps) * ln_w[e,n] + ln_b[e,n]
__global__ void ln_apply_kernel(float* __restrict__ state,
                                const float* __restrict__ h,
                                const float* __restrict__ lnw,
                                const float* __restrict__ lnb,
                                const double* __restrict__ stats,
                                int rowsPerBatch) {
  long long cnt = (long long)rowsPerBatch * 256;
  long long t = (long long)blockIdx.x * 256 + threadIdx.x;
  if (t >= cnt * BS_) return;
  int b = (int)(t / cnt);
  long long rem = t - (long long)b * cnt;
  double mu = stats[2 * b] / (double)cnt;
  double var = stats[2 * b + 1] / (double)cnt - mu * mu;
  float rs = rsqrtf((float)var + 1e-5f);
  state[t] += (h[t] - (float)mu) * rs * lnw[rem] + lnb[rem];
}

// ---------------------------------------------------------------------------
static inline int cdiv(long long a, long long b) { return (int)((a + b - 1) / b); }

extern "C" void kernel_launch(void* const* d_in, const int* in_sizes, int n_in,
                              void* d_out, int out_size, void* d_ws, size_t ws_size,
                              hipStream_t stream) {
  (void)in_sizes; (void)n_in; (void)out_size; (void)ws_size;
  const float* gx_in = (const float*)d_in[0];
  const float* mx_in = (const float*)d_in[1];
  const int*   me_i  = (const int*)d_in[2];
  const float* me_x  = (const float*)d_in[3];
  const int*   g2m_i = (const int*)d_in[4];
  const float* g2m_x = (const float*)d_in[5];
  const int*   m2g_i = (const int*)d_in[6];
  const float* m2g_x = (const float*)d_in[7];
  const float* gs    = (const float*)d_in[8];
  const float* lsm   = (const float*)d_in[9];
  auto P = [&](int i) { return (const float*)d_in[i]; };

  // workspace carve (f32), 256B aligned
  char* w = (char*)d_ws; size_t off = 0;
  auto alloc = [&](size_t elems) {
    float* p = (float*)(w + off);
    off = (off + elems * sizeof(float) + 255) & ~(size_t)255;
    return p;
  };
  float* gxs   = alloc((size_t)BS_ * GNUM_ * 256);
  float* mxs   = alloc((size_t)BS_ * MNUM_ * 256);
  float* mes   = alloc((size_t)BS_ * ME_E  * 256);
  float* g2mes = alloc((size_t)BS_ * G2M_E * 256);
  float* m2ges = alloc((size_t)BS_ * M2G_E * 256);
  float* h1    = alloc((size_t)BS_ * M2G_E * 512);
  float* h2    = alloc((size_t)BS_ * M2G_E * 256);
  float* h3    = alloc((size_t)BS_ * M2G_E * 256);
  float* agg   = alloc((size_t)BS_ * GNUM_ * 256);
  double* stats = (double*)(w + off); off += 256;

  GSrc nil{nullptr, nullptr, 0, 0};
  auto gemm = [&](GSrc a, GSrc b, GSrc c, const float* Wm, const float* bi,
                  float* o, int M, int N, int K, int rpb, int ldo, int act) {
    dim3 g(cdiv(M, 64), cdiv(N, 128));
    wmma_mlp_kernel<<<g, 256, 0, stream>>>(a, b, c, Wm, bi, o, M, N, K, rpb, ldo, act);
  };
  auto lnres = [&](float* state, const float* lnw, const float* lnb, int rpb) {
    zero_kernel<<<1, 64, 0, stream>>>((float*)stats, 4 * BS_);
    long long cnt = (long long)rpb * 256;
    int blk = cdiv(cnt, 256 * 32); if (blk > 2048) blk = 2048;
    ln_stats_kernel<<<dim3(blk, BS_), 256, 0, stream>>>(h3, rpb, stats);
    ln_apply_kernel<<<cdiv(cnt * BS_, 256), 256, 0, stream>>>(state, h3, lnw, lnb, stats, rpb);
  };
  auto mlp3 = [&](GSrc a, GSrc b, GSrc c, int K, int base, int rpb, float* state) {
    int M = BS_ * rpb;
    gemm(a, b, c, P(base + 0), P(base + 1), h1, M, 512, K, rpb, 512, 1);
    gemm({h1, nullptr, rpb, 512}, nil, nil, P(base + 2), P(base + 3), h2, M, 256, 512, rpb, 256, 1);
    gemm({h2, nullptr, rpb, 256}, nil, nil, P(base + 4), P(base + 5), h3, M, 256, 256, rpb, 256, 0);
    lnres(state, P(base + 7), P(base + 6), rpb);   // ln_w = +7, ln_b = +6
  };
  auto scatter = [&](float* dst, const float* src, const int* col, int E, int dstRows) {
    long long zn = (long long)BS_ * dstRows * 256;
    zero_kernel<<<cdiv(zn, 256), 256, 0, stream>>>(dst, zn);
    long long tn = (long long)BS_ * E * 256;
    scatter_add_kernel<<<cdiv(tn, 256), 256, 0, stream>>>(dst, src, col, E, dstRows);
  };

  // ---- embeddings ----
  emb_grid_kernel<<<BS_ * GNUM_, 256, 0, stream>>>(gx_in, gs, lsm, P(10), P(11), gxs);
  small_emb_kernel<<<cdiv((long long)MNUM_ * 256, 256), 256, 0, stream>>>(mx_in, P(14), P(15), mxs, MNUM_, 3, BS_);
  small_emb_kernel<<<cdiv((long long)ME_E  * 256, 256), 256, 0, stream>>>(me_x,  P(18), P(19), mes,  ME_E,  4, BS_);
  small_emb_kernel<<<cdiv((long long)G2M_E * 256, 256), 256, 0, stream>>>(g2m_x, P(12), P(13), g2mes, G2M_E, 4, BS_);
  small_emb_kernel<<<cdiv((long long)M2G_E * 256, 256), 256, 0, stream>>>(m2g_x, P(16), P(17), m2ges, M2G_E, 4, BS_);

  // ---- Grid2Mesh edge + node ----
  mlp3({gxs, g2m_i, GNUM_, 256}, {mxs, g2m_i + G2M_E, MNUM_, 256},
       {g2mes, nullptr, G2M_E, 256}, 768, 20, G2M_E, g2mes);
  scatter(agg, g2mes, g2m_i + G2M_E, G2M_E, MNUM_);
  mlp3({mxs, nullptr, MNUM_, 256}, {agg, nullptr, MNUM_, 256}, nil, 512, 34, MNUM_, mxs);
  // grid residual MLP (g2m_grid)
  gemm({gxs, nullptr, GNUM_, 256}, nil, nil, P(28), P(29), h2, BS_ * GNUM_, 256, 256, GNUM_, 256, 1);
  gemm({h2, nullptr, GNUM_, 256}, nil, nil, P(30), P(31), h3, BS_ * GNUM_, 256, 256, GNUM_, 256, 0);
  lnres(gxs, P(33), P(32), GNUM_);

  // ---- 2x mesh processor ----
  for (int st = 0; st < 2; ++st) {
    int eb = 64 + st * 8, nb = 80 + st * 8;
    mlp3({mxs, me_i, MNUM_, 256}, {mxs, me_i + ME_E, MNUM_, 256},
         {mes, nullptr, ME_E, 256}, 768, eb, ME_E, mes);
    scatter(agg, mes, me_i + ME_E, ME_E, MNUM_);
    mlp3({mxs, nullptr, MNUM_, 256}, {agg, nullptr, MNUM_, 256}, nil, 512, nb, MNUM_, mxs);
  }

  // ---- Mesh2Grid edge + node ----
  mlp3({mxs, m2g_i, MNUM_, 256}, {gxs, m2g_i + M2G_E, GNUM_, 256},
       {m2ges, nullptr, M2G_E, 256}, 768, 48, M2G_E, m2ges);
  scatter(agg, m2ges, m2g_i + M2G_E, M2G_E, GNUM_);
  mlp3({gxs, nullptr, GNUM_, 256}, {agg, nullptr, GNUM_, 256}, nil, 512, 56, GNUM_, gxs);

  // ---- head: 256 -> 256 -> 128 -> 70 ----
  gemm({gxs, nullptr, GNUM_, 256}, nil, nil, P(42), P(43), h2, BS_ * GNUM_, 256, 256, GNUM_, 256, 1);
  gemm({h2,  nullptr, GNUM_, 256}, nil, nil, P(44), P(45), h1, BS_ * GNUM_, 128, 256, GNUM_, 128, 1);
  gemm({h1,  nullptr, GNUM_, 128}, nil, nil, P(46), P(47), (float*)d_out,
       BS_ * GNUM_, 70, 128, GNUM_, 70, 0);
}
